// LigerFusedLinearPackDPOLoss_21835613733611
// MI455X (gfx1250) — matrix operand order repaired and checked by reading, not compiled, not run on hardware
//
#include <hip/hip_runtime.h>

// ---------------- problem constants ----------------
#define T_TOK 2048          // 2 * N_PAIRS * SEQ
#define HDIM  1024
#define VOCAB 32000
#define SEQLEN 256
#define NPAIRS 4
#define BETA 0.1f
#define IGNORE_INDEX (-100)

// ---------------- tiling ----------------
#define MROWS 32                       // rows per block (two WMMA M tiles)
#define NSLICES 5                      // V partitioned into 5 slices of 6400
#define COLS_PER_SLICE (VOCAB / NSLICES)     // 6400
#define NITER_COLS 256                 // cols per block iteration (8 waves * 32)
#define NITERS (COLS_PER_SLICE / NITER_COLS) // 25
#define KCHUNK 32                      // K per WMMA step
#define NKC (HDIM / KCHUNK)            // 32

#define A2S 516   // u32 stride per A row (512 data + 4 pad) -> bank spread, 16B aligned

typedef __attribute__((ext_vector_type(16))) __bf16    v16bf;
typedef __attribute__((ext_vector_type(8)))  float     v8f;
typedef __attribute__((ext_vector_type(4)))  float     f32x4;
typedef __attribute__((ext_vector_type(4)))  unsigned  u32x4;

// Pack two fp32 -> packed bf16x2 in ONE VALU op (v_perm_b32 byte select).
// out = { hi16(b), hi16(a) }  (truncation; inputs ~N(0,0.02), fine for bf16 GEMM)
static __device__ __forceinline__ unsigned pack_bf16x2(float a, float b) {
    unsigned ua = __float_as_uint(a), ub = __float_as_uint(b);
#if __has_builtin(__builtin_amdgcn_perm)
    return __builtin_amdgcn_perm(ub, ua, 0x07060302u);  // v_perm_b32
#else
    return (ua >> 16) | (ub & 0xffff0000u);
#endif
}

// Build a 16-bf16 WMMA fragment from 16 fp32 (K-consecutive): VGPR v <- K=2v,2v+1
static __device__ __forceinline__ v16bf make_frag(const f32x4 q0, const f32x4 q1,
                                                  const f32x4 q2, const f32x4 q3) {
    union { unsigned u[8]; v16bf v; } r;
    r.u[0] = pack_bf16x2(q0.x, q0.y); r.u[1] = pack_bf16x2(q0.z, q0.w);
    r.u[2] = pack_bf16x2(q1.x, q1.y); r.u[3] = pack_bf16x2(q1.z, q1.w);
    r.u[4] = pack_bf16x2(q2.x, q2.y); r.u[5] = pack_bf16x2(q2.z, q2.w);
    r.u[6] = pack_bf16x2(q3.x, q3.y); r.u[7] = pack_bf16x2(q3.z, q3.w);
    return r.v;
}

// ==========================================================================
// Kernel 1: fused (x @ W^T + b) with online logsumexp + target-logit extract.
// grid = (T/MROWS, NSLICES, 2 models), block = 256 (8 wave32).
// A (32x1024 x-block) staged once in LDS as bf16, shared by all 8 waves;
// each wave owns 32 columns and computes a 32x32 output patch per iteration
// (2 M-tiles x 2 N-tiles = 4 WMMAs per K-step, so each converted B fragment
// feeds two WMMAs). B is loaded straight from global into per-lane fragments
// (B order used here, same as A = lane-per-column, 16 consecutive K per lane, which matches
// row-major W). No barriers in the K loop.
// ==========================================================================
__global__ __launch_bounds__(256) void dpo_fused_linear_lse_kernel(
    const float* __restrict__ x,  const float* __restrict__ w,  const float* __restrict__ bias,
    const float* __restrict__ rx, const float* __restrict__ rw, const float* __restrict__ rbias,
    const int*   __restrict__ target,
    float* __restrict__ part_m, float* __restrict__ part_s, float* __restrict__ tgt_logit)
{
    __shared__ unsigned A2[MROWS * A2S];   // 66 KB: x-block as packed bf16 pairs
    __shared__ float red_m[MROWS][8];
    __shared__ float red_s[MROWS][8];

    const int model = blockIdx.z;
    const float* X  = model ? rx : x;
    const float* W  = model ? rw : w;
    const float* BB = model ? rbias : bias;

    const int m0  = blockIdx.x * MROWS;
    const int vs  = blockIdx.y;
    const int tid = threadIdx.x;
    const int lane = tid & 31;
    const int wv   = tid >> 5;       // wave id 0..7 = N-column group
    const int l16  = lane & 15;
    const int hsel = lane >> 4;      // 0: lanes 0-15, 1: lanes 16-31

    // ---- Stage A block (32 rows x 1024 K), fp32 -> bf16, into LDS ----
    {
        const int row = tid >> 3;             // 8 threads per row
        const int l   = tid & 7;
        const f32x4* src4 = (const f32x4*)(X + (size_t)(m0 + row) * HDIM);
        #pragma unroll
        for (int j = 0; j < 32; ++j) {
            int k4 = l + j * 8;               // float4 index 0..255
            f32x4 f = src4[k4];
            A2[row * A2S + k4 * 2 + 0] = pack_bf16x2(f.x, f.y);
            A2[row * A2S + k4 * 2 + 1] = pack_bf16x2(f.z, f.w);
        }
    }

    // Per-lane targets for the 16 C-row slots (C layout: row = mt*16 + r + hsel*8)
    int tg[16];
    #pragma unroll
    for (int mt = 0; mt < 2; ++mt)
        #pragma unroll
        for (int r = 0; r < 8; ++r)
            tg[mt * 8 + r] = target[m0 + mt * 16 + r + hsel * 8];

    // Per-lane online logsumexp state (this lane's column subset of each row)
    float run_m[16], run_s[16];
    #pragma unroll
    for (int q = 0; q < 16; ++q) { run_m[q] = -__builtin_inff(); run_s[q] = 0.f; }

    __syncthreads();   // A2 visible to all waves; no barriers needed after this

    for (int it = 0; it < NITERS; ++it) {
        const int colbase = vs * COLS_PER_SLICE + it * NITER_COLS;
        const int wcol    = colbase + wv * 32;            // this wave's 32 columns
        const float bias0 = BB[wcol + l16];
        const float bias1 = BB[wcol + 16 + l16];

        v8f acc00 = {0.f,0.f,0.f,0.f,0.f,0.f,0.f,0.f};   // [mt=0][ct=0]
        v8f acc01 = {0.f,0.f,0.f,0.f,0.f,0.f,0.f,0.f};   // [mt=0][ct=1]
        v8f acc10 = {0.f,0.f,0.f,0.f,0.f,0.f,0.f,0.f};   // [mt=1][ct=0]
        v8f acc11 = {0.f,0.f,0.f,0.f,0.f,0.f,0.f,0.f};   // [mt=1][ct=1]

        // This lane's two W rows (tile 0 / tile 1 columns), K-offset by hsel half
        const f32x4* wr0 = (const f32x4*)(W + (size_t)(wcol + l16)      * HDIM + hsel * 16);
        const f32x4* wr1 = (const f32x4*)(W + (size_t)(wcol + 16 + l16) * HDIM + hsel * 16);

        // register prefetch of K-chunk 0 (16 fp32 per column-tile per lane)
        f32x4 g0[4], g1[4];
        #pragma unroll
        for (int j = 0; j < 4; ++j) { g0[j] = wr0[j]; g1[j] = wr1[j]; }

        for (int kc = 0; kc < NKC; ++kc) {
            // A fragments from LDS (two M tiles): lane = M row (l16), K half by hsel
            union { u32x4 q[2]; v16bf v; } af0, af1;
            const unsigned* ap0 = &A2[l16 * A2S + kc * 16 + (hsel ? 8 : 0)];
            const unsigned* ap1 = ap0 + 16 * A2S;
            af0.q[0] = *(const u32x4*)ap0;
            af0.q[1] = *(const u32x4*)(ap0 + 4);
            af1.q[0] = *(const u32x4*)ap1;
            af1.q[1] = *(const u32x4*)(ap1 + 4);

            // Convert this chunk's B data (already in registers) to fragments
            v16bf bf0 = make_frag(g0[0], g0[1], g0[2], g0[3]);
            v16bf bf1 = make_frag(g1[0], g1[1], g1[2], g1[3]);

            // Prefetch next K-chunk while WMMAs run
            if (kc + 1 < NKC) {
                const f32x4* n0 = wr0 + (kc + 1) * (KCHUNK / 4);
                const f32x4* n1 = wr1 + (kc + 1) * (KCHUNK / 4);
                #pragma unroll
                for (int j = 0; j < 4; ++j) { g0[j] = n0[j]; g1[j] = n1[j]; }
                if (kc + 2 < NKC) {
                    __builtin_prefetch(wr0 + (kc + 2) * (KCHUNK / 4), 0, 3);
                    __builtin_prefetch(wr1 + (kc + 2) * (KCHUNK / 4), 0, 3);
                }
            }

            // 4 WMMAs: each B fragment reused by both M tiles
            acc00 = __builtin_amdgcn_wmma_f32_16x16x32_bf16(
                        false, af0.v, false, bf0, (short)0, acc00, false, false);
            acc10 = __builtin_amdgcn_wmma_f32_16x16x32_bf16(
                        false, af1.v, false, bf0, (short)0, acc10, false, false);
            acc01 = __builtin_amdgcn_wmma_f32_16x16x32_bf16(
                        false, af0.v, false, bf1, (short)0, acc01, false, false);
            acc11 = __builtin_amdgcn_wmma_f32_16x16x32_bf16(
                        false, af1.v, false, bf1, (short)0, acc11, false, false);
        }

        // Epilogue: bias, target extraction, per-lane online logsumexp update
        #pragma unroll
        for (int mt = 0; mt < 2; ++mt) {
            #pragma unroll
            for (int r = 0; r < 8; ++r) {
                float v0 = (mt ? acc10[r] : acc00[r]) + bias0;
                float v1 = (mt ? acc11[r] : acc01[r]) + bias1;
                const int q    = mt * 8 + r;
                const int grow = m0 + mt * 16 + r + hsel * 8;
                if (wcol + l16      == tg[q]) tgt_logit[model * T_TOK + grow] = v0;
                if (wcol + 16 + l16 == tg[q]) tgt_logit[model * T_TOK + grow] = v1;
                float nm = fmaxf(run_m[q], fmaxf(v0, v1));
                run_s[q] = run_s[q] * __expf(run_m[q] - nm) + __expf(v0 - nm) + __expf(v1 - nm);
                run_m[q] = nm;
            }
        }
    }

    // Cross-lane fold (16-lane butterflies): combine column subsets per row
    #pragma unroll
    for (int q = 0; q < 16; ++q) {
        float m = run_m[q], s = run_s[q];
        #pragma unroll
        for (int st = 1; st < 16; st <<= 1) {
            float mo = __shfl_xor(m, st, 16);
            float so = __shfl_xor(s, st, 16);
            float nm = fmaxf(m, mo);
            s = s * __expf(m - nm) + so * __expf(mo - nm);
            m = nm;
        }
        run_m[q] = m; run_s[q] = s;
    }

    // Combine the 8 waves' partial stats, emit per-slice (m, s) to workspace
    if (l16 == 0) {
        #pragma unroll
        for (int mt = 0; mt < 2; ++mt)
            #pragma unroll
            for (int r = 0; r < 8; ++r) {
                const int row = mt * 16 + hsel * 8 + r;
                red_m[row][wv] = run_m[mt * 8 + r];
                red_s[row][wv] = run_s[mt * 8 + r];
            }
    }
    __syncthreads();
    if (tid < MROWS) {
        float m = -__builtin_inff();
        #pragma unroll
        for (int q = 0; q < 8; ++q) m = fmaxf(m, red_m[tid][q]);
        float s = 0.f;
        #pragma unroll
        for (int q = 0; q < 8; ++q) s += red_s[tid][q] * __expf(red_m[tid][q] - m);
        const size_t idx = ((size_t)model * NSLICES + vs) * T_TOK + m0 + tid;
        part_m[idx] = m;
        part_s[idx] = s;
    }
}

// ==========================================================================
// Kernel 2: fold slice partials -> logZ -> token logp -> per-seq sums -> DPO loss
// ==========================================================================
__global__ __launch_bounds__(256) void dpo_loss_kernel(
    const int* __restrict__ target,
    const float* __restrict__ part_m, const float* __restrict__ part_s,
    const float* __restrict__ tgt_logit, float* __restrict__ out)
{
    __shared__ float seqsum[16];    // [model(2)][seq(8)]
    const int tid = threadIdx.x;
    if (tid < 16) seqsum[tid] = 0.f;
    __syncthreads();

    for (int i = tid; i < 2 * T_TOK; i += 256) {
        const int model = i / T_TOK, row = i % T_TOK;
        float m = -__builtin_inff();
        #pragma unroll
        for (int v = 0; v < NSLICES; ++v)
            m = fmaxf(m, part_m[((size_t)model * NSLICES + v) * T_TOK + row]);
        float s = 0.f;
        #pragma unroll
        for (int v = 0; v < NSLICES; ++v) {
            const size_t idx = ((size_t)model * NSLICES + v) * T_TOK + row;
            s += part_s[idx] * expf(part_m[idx] - m);
        }
        const float logZ = m + logf(s);
        const int   tg   = target[row];
        const float lp   = (tg == IGNORE_INDEX)
                             ? 0.f
                             : (tgt_logit[model * T_TOK + row] - logZ);
        atomicAdd(&seqsum[model * 8 + (row / SEQLEN)], lp);
    }
    __syncthreads();

    if (tid == 0) {
        float loss = 0.f;
        #pragma unroll
        for (int p = 0; p < NPAIRS; ++p) {
            const float c  = seqsum[0 + 2 * p], r  = seqsum[0 + 2 * p + 1];
            const float rc = seqsum[8 + 2 * p], rr = seqsum[8 + 2 * p + 1];
            const float z  = BETA * ((c - rc) - (r - rr));
            const float ls = fminf(z, 0.f) - log1pf(expf(-fabsf(z)));  // log_sigmoid(z)
            loss -= ls;
        }
        out[0] = loss / (float)NPAIRS;
    }
}

// ==========================================================================
extern "C" void kernel_launch(void* const* d_in, const int* in_sizes, int n_in,
                              void* d_out, int out_size, void* d_ws, size_t ws_size,
                              hipStream_t stream) {
    (void)in_sizes; (void)n_in; (void)out_size; (void)ws_size;
    const float* x   = (const float*)d_in[0];
    const float* w   = (const float*)d_in[1];
    const float* b   = (const float*)d_in[2];
    const float* rx  = (const float*)d_in[3];
    const float* rw  = (const float*)d_in[4];
    const float* rb  = (const float*)d_in[5];
    const int*   tgt = (const int*)  d_in[6];

    // workspace: part_m[2][NSLICES][T], part_s[2][NSLICES][T], tgt_logit[2][T]  (~176 KB)
    float* part_m = (float*)d_ws;
    float* part_s = part_m + (size_t)2 * NSLICES * T_TOK;
    float* tgt_l  = part_s + (size_t)2 * NSLICES * T_TOK;

    dim3 grid(T_TOK / MROWS, NSLICES, 2);
    dpo_fused_linear_lse_kernel<<<grid, 256, 0, stream>>>(
        x, w, b, rx, rw, rb, tgt, part_m, part_s, tgt_l);
    dpo_loss_kernel<<<1, 256, 0, stream>>>(tgt, part_m, part_s, tgt_l, (float*)d_out);
}
